// Decoder_34875134443572
// MI455X (gfx1250) — matrix operand order repaired and checked by reading, not compiled, not run on hardware
//
#include <hip/hip_runtime.h>
#include <hip/hip_bf16.h>
#include <math.h>

// ---------------------------------------------------------------------------
// Image-to-LaTeX attention decoder for MI455X (gfx1250), wave32 + WMMA f16.
// GEMMs: v_wmma_f32_16x16x32_f16, fragment-order pre-packed f16 weights
// (one contiguous v16h load per lane), float4-vectorized A loads, dual
// independent accumulators to fill the WMMA hazard window.
// ---------------------------------------------------------------------------

typedef __attribute__((ext_vector_type(16))) _Float16 v16h;
typedef __attribute__((ext_vector_type(8)))  float    v8f;
typedef __attribute__((ext_vector_type(4)))  float    v4f;

#define WMMA_F16(a, b, c) \
  __builtin_amdgcn_wmma_f32_16x16x32_f16(false, (a), false, (b), (short)0, (c), false, false)

__device__ __forceinline__ float sigmoidf_(float x) { return 1.0f / (1.0f + __expf(-x)); }

// A fragment: 16x32 tile, fp32 source, row-major, lda multiple of 8, rows
// 16B-aligned. ISA layout: lane l, r=l&15, h=l>>4; elements map to row floats
// [h*8 .. h*8+7] and [16+h*8 .. 16+h*8+7] -> four aligned float4 loads.
__device__ __forceinline__ v16h frag_a_f32(const float* __restrict__ A, int lda) {
  const int lane = threadIdx.x & 31;
  const int r    = lane & 15;
  const int hh   = (lane >> 4) & 1;
  const float* row = A + (size_t)r * lda + hh * 8;
  const v4f f0 = *(const v4f*)(row);
  const v4f f1 = *(const v4f*)(row + 4);
  const v4f f2 = *(const v4f*)(row + 16);
  const v4f f3 = *(const v4f*)(row + 20);
  v16h a;
#pragma unroll
  for (int i = 0; i < 4; ++i) {
    a[i]      = (_Float16)f0[i];
    a[4 + i]  = (_Float16)f1[i];
    a[8 + i]  = (_Float16)f2[i];
    a[12 + i] = (_Float16)f3[i];
  }
  return a;
}

// B fragment from fragment-order packed weights: tile (kt,nt) is 512
// contiguous f16, lane-major [32][16] -> one 32B contiguous load per lane.
__device__ __forceinline__ v16h frag_b_packed(const _Float16* __restrict__ tile) {
  const int lane = threadIdx.x & 31;
  return *(const v16h*)(tile + (size_t)lane * 16);
}

// ---------------------------------------------------------------------------
// Generic GEMM (1 tile/wave): C[M,16*gridDim.x] = A[M,K](f32) x Bp + bias.
// grid = (nTiles, M/16), block = 32. Used where wave-parallelism matters
// more than per-wave efficiency (small dec-projection GEMM).
// ---------------------------------------------------------------------------
__global__ void k_gemm_wmma(const float* __restrict__ A, int lda,
                            const _Float16* __restrict__ Bp, int nTiles,
                            const float* __restrict__ bias,
                            float* __restrict__ C, int ldc, int K) {
  const int nt    = blockIdx.x;
  const int mBase = blockIdx.y * 16;
  const size_t kStride = (size_t)nTiles * 512;  // packed elems per k-tile row
  const _Float16* tile = Bp + (size_t)nt * 512;
  v8f acc = {};
  for (int k0 = 0; k0 < K; k0 += 32, tile += kStride) {
    if (k0 + 32 < K) __builtin_prefetch(tile + kStride, 0, 3);  // WGP-scope
    v16h a = frag_a_f32(A + (size_t)mBase * lda + k0, lda);
    v16h b = frag_b_packed(tile);
    acc = WMMA_F16(a, b, acc);
  }
  const int lane = threadIdx.x & 31;
  const int n    = lane & 15;
  const int hh   = (lane >> 4) & 1;
  const float bv = bias ? bias[nt * 16 + n] : 0.0f;
#pragma unroll
  for (int r = 0; r < 8; ++r)
    C[(size_t)(mBase + hh * 8 + r) * ldc + nt * 16 + n] = acc[r] + bv;
}

// ---------------------------------------------------------------------------
// Dual-M GEMM (2 m-tiles/wave, shared B fragment): the second WMMA is
// independent of the first, filling the 4-coexec WMMA hazard window, and B
// fragment loads amortize 2x. grid = (nTiles, M/32), block = 32.
// ---------------------------------------------------------------------------
__global__ void k_gemm_wmma2(const float* __restrict__ A, int lda,
                             const _Float16* __restrict__ Bp, int nTiles,
                             const float* __restrict__ bias,
                             float* __restrict__ C, int ldc, int K) {
  const int nt    = blockIdx.x;
  const int mBase = blockIdx.y * 32;
  const size_t kStride = (size_t)nTiles * 512;
  const _Float16* tile = Bp + (size_t)nt * 512;
  v8f acc0 = {}, acc1 = {};
  for (int k0 = 0; k0 < K; k0 += 32, tile += kStride) {
    if (k0 + 32 < K) __builtin_prefetch(tile + kStride, 0, 3);
    v16h a0 = frag_a_f32(A + (size_t)mBase * lda + k0, lda);
    v16h a1 = frag_a_f32(A + (size_t)(mBase + 16) * lda + k0, lda);
    v16h b  = frag_b_packed(tile);
    acc0 = WMMA_F16(a0, b, acc0);
    acc1 = WMMA_F16(a1, b, acc1);  // independent: fills hazard slots
  }
  const int lane = threadIdx.x & 31;
  const int n    = lane & 15;
  const int hh   = (lane >> 4) & 1;
  const float bv = bias ? bias[nt * 16 + n] : 0.0f;
#pragma unroll
  for (int r = 0; r < 8; ++r) {
    C[(size_t)(mBase + hh * 8 + r) * ldc + nt * 16 + n]      = acc0[r] + bv;
    C[(size_t)(mBase + 16 + hh * 8 + r) * ldc + nt * 16 + n] = acc1[r] + bv;
  }
}

// Logits GEMM: dual-M, packed B padded to 1008 cols, only cols < Nvalid
// stored; C strides into preds[B,S,V]. grid = (63, 4), block = 32.
__global__ void k_gemm_logits(const float* __restrict__ A, int lda,
                              const _Float16* __restrict__ Bp, int nTiles,
                              const float* __restrict__ bias,
                              float* __restrict__ C, size_t ldc, int K, int Nvalid) {
  const int nt    = blockIdx.x;
  const int mBase = blockIdx.y * 32;
  const size_t kStride = (size_t)nTiles * 512;
  const _Float16* tile = Bp + (size_t)nt * 512;
  v8f acc0 = {}, acc1 = {};
  for (int k0 = 0; k0 < K; k0 += 32, tile += kStride) {
    if (k0 + 32 < K) __builtin_prefetch(tile + kStride, 0, 3);
    v16h a0 = frag_a_f32(A + (size_t)mBase * lda + k0, lda);
    v16h a1 = frag_a_f32(A + (size_t)(mBase + 16) * lda + k0, lda);
    v16h b  = frag_b_packed(tile);
    acc0 = WMMA_F16(a0, b, acc0);
    acc1 = WMMA_F16(a1, b, acc1);
  }
  const int lane = threadIdx.x & 31;
  const int n    = lane & 15;
  const int hh   = (lane >> 4) & 1;
  const int col  = nt * 16 + n;
  if (col < Nvalid) {
    const float bv = bias[col];
#pragma unroll
    for (int r = 0; r < 8; ++r) {
      C[(size_t)(mBase + hh * 8 + r) * ldc + col]      = acc0[r] + bv;
      C[(size_t)(mBase + 16 + hh * 8 + r) * ldc + col] = acc1[r] + bv;
    }
  }
}

// ---------------------------------------------------------------------------
// Fused LSTM step: gates = [A0|A1] x W(packed f16, 128 n-tiles over 4D=2048)
// + bias, then cell update. Each wave owns one 16x16 (m,d) tile across all 4
// gates (4 independent back-to-back WMMAs -> hazard window stays full), so
// the epilogue does the whole cell update. grid=(32, 8), block=32.
// ---------------------------------------------------------------------------
__global__ void k_lstm_step(const float* __restrict__ A0, int lda0, int K0,
                            const float* __restrict__ A1, int lda1, int K1,
                            const _Float16* __restrict__ Wp,   // packed, nTiles=128
                            const float* __restrict__ bias,    // [2048]
                            float* __restrict__ cbuf,          // [128,512] in/out
                            float* __restrict__ hout) {        // [128,512] out
  const int nt    = blockIdx.x;       // d tile within D=512 (0..31)
  const int mBase = blockIdx.y * 16;  // batch tile
  const size_t kStride = (size_t)128 * 512;
  v8f acc[4] = {v8f{}, v8f{}, v8f{}, v8f{}};
  const int K = K0 + K1;
  for (int k0 = 0; k0 < K; k0 += 32) {
    v16h a = (k0 < K0) ? frag_a_f32(A0 + (size_t)mBase * lda0 + k0, lda0)
                       : frag_a_f32(A1 + (size_t)mBase * lda1 + (k0 - K0), lda1);
    const _Float16* ktile = Wp + (size_t)(k0 >> 5) * kStride;
    if (k0 + 32 < K) __builtin_prefetch(ktile + kStride + (size_t)nt * 512, 0, 3);
#pragma unroll
    for (int g = 0; g < 4; ++g) {
      v16h b = frag_b_packed(ktile + (size_t)(g * 32 + nt) * 512);
      acc[g] = WMMA_F16(a, b, acc[g]);
    }
  }
  const int lane = threadIdx.x & 31;
  const int n    = nt * 16 + (lane & 15);
  const int hh   = (lane >> 4) & 1;
  const float bi = bias[n], bff = bias[512 + n], bgg = bias[1024 + n], bo = bias[1536 + n];
#pragma unroll
  for (int r = 0; r < 8; ++r) {
    const int m    = mBase + hh * 8 + r;
    const float gi = sigmoidf_(acc[0][r] + bi);
    const float gf = sigmoidf_(acc[1][r] + bff);
    const float gg = tanhf(acc[2][r] + bgg);
    const float go = sigmoidf_(acc[3][r] + bo);
    const float cn = gf * cbuf[(size_t)m * 512 + n] + gi * gg;
    cbuf[(size_t)m * 512 + n] = cn;
    hout[(size_t)m * 512 + n] = go * tanhf(cn);
  }
}

// ---------------------------------------------------------------------------
// Attention (+ fused embedding gather): one block per batch row.
// scores -> softmax -> alpha (to d_out) and awe = alpha . features.
// ---------------------------------------------------------------------------
__global__ void k_attention(const float* __restrict__ enc_t,    // [128,196,512]
                            const float* __restrict__ dec_t,    // [128,512]
                            const float* __restrict__ features, // [128,196,1024]
                            const float* __restrict__ wa,       // [512]
                            const float* __restrict__ battn,    // [1]
                            const float* __restrict__ emb,      // [1000,128]
                            const int* __restrict__ latexes,    // [128,257]
                            int s,
                            float* __restrict__ alpha_out,      // + s*196 already
                            size_t alpha_bstride,               // 256*196
                            float* __restrict__ xea) {          // [128,1152]
  __shared__ float sdec[512];
  __shared__ float sc[256];
  __shared__ float tmp[256];
  const int b   = blockIdx.x;
  const int tid = threadIdx.x;  // 256 threads = 8 waves

  if (tid < 128) {  // fused embedding gather -> xea[:,0:128]
    const int tok = latexes[(size_t)b * 257 + s];
    xea[(size_t)b * 1152 + tid] = emb[(size_t)tok * 128 + tid];
  }
  for (int i = tid; i < 512; i += 256) sdec[i] = dec_t[(size_t)b * 512 + i];
  __syncthreads();

  const int wv = tid >> 5, lane = tid & 31;
  for (int p = wv; p < 196; p += 8) {
    const float* e = enc_t + ((size_t)b * 196 + p) * 512;
    float sum = 0.0f;
    for (int a = lane; a < 512; a += 32) sum += tanhf(e[a] + sdec[a]) * wa[a];
#pragma unroll
    for (int off = 16; off > 0; off >>= 1) sum += __shfl_down(sum, off, 32);
    if (lane == 0) sc[p] = sum + battn[0];
  }
  __syncthreads();

  const float v = (tid < 196) ? sc[tid] : -INFINITY;
  tmp[tid] = v;
  __syncthreads();
  for (int s2 = 128; s2 > 0; s2 >>= 1) {
    if (tid < s2) tmp[tid] = fmaxf(tmp[tid], tmp[tid + s2]);
    __syncthreads();
  }
  const float mx = tmp[0];
  __syncthreads();
  const float ex = (tid < 196) ? __expf(v - mx) : 0.0f;
  tmp[tid] = ex;
  __syncthreads();
  for (int s2 = 128; s2 > 0; s2 >>= 1) {
    if (tid < s2) tmp[tid] += tmp[tid + s2];
    __syncthreads();
  }
  const float inv = 1.0f / tmp[0];
  __syncthreads();
  if (tid < 196) {
    const float al = ex * inv;
    sc[tid] = al;
    alpha_out[(size_t)b * alpha_bstride + tid] = al;
  }
  __syncthreads();

  for (int e0 = tid; e0 < 1024; e0 += 256) {
    float sum = 0.0f;
    const float* f = features + (size_t)b * 196 * 1024 + e0;
    for (int p = 0; p < 196; ++p) sum += sc[p] * f[(size_t)p * 1024];
    xea[(size_t)b * 1152 + 128 + e0] = sum;
  }
}

// ---------------------------------------------------------------------------
// Setup helpers
// ---------------------------------------------------------------------------
__global__ void k_mean(const float* __restrict__ features, float* __restrict__ mean_enc) {
  const int b = blockIdx.x, tid = threadIdx.x;
  for (int e = tid; e < 1024; e += 256) {
    float s = 0.0f;
    const float* f = features + (size_t)b * 196 * 1024 + e;
    for (int p = 0; p < 196; ++p) s += f[(size_t)p * 1024];
    mean_enc[(size_t)b * 1024 + e] = s * (1.0f / 196.0f);
  }
}

__global__ void k_init_state(const float* __restrict__ mean_enc,
                             const float* __restrict__ Wh0, const float* __restrict__ bh0,
                             const float* __restrict__ Wc0, const float* __restrict__ bc0,
                             float* __restrict__ h1, float* __restrict__ c1,
                             float* __restrict__ h2, float* __restrict__ c2) {
  __shared__ float sm[1024];
  const int b = blockIdx.x, tid = threadIdx.x;
  for (int e = tid; e < 1024; e += 256) sm[e] = mean_enc[(size_t)b * 1024 + e];
  __syncthreads();
  for (int d = tid; d < 512; d += 256) {
    float sh = bh0[d], sc = bc0[d];
    const float* wh = Wh0 + (size_t)d * 1024;
    const float* wc = Wc0 + (size_t)d * 1024;
    for (int e = 0; e < 1024; ++e) { sh += wh[e] * sm[e]; sc += wc[e] * sm[e]; }
    h1[(size_t)b * 512 + d] = sh; c1[(size_t)b * 512 + d] = sc;
    h2[(size_t)b * 512 + d] = sh; c2[(size_t)b * 512 + d] = sc;
  }
}

// Pack W[N,K] fp32 -> fragment-order f16 tiles. Dest tile (ktOff+kt, nt) at
// ((ktOff+kt)*nTiles + nt)*512; element (l,e): n = nt*16 + (l&15),
// k = kt*32 + (e>>3)*16 + (l>>4)*8 + (e&7). Columns n >= N are zero-filled (vocab padding).
__global__ void k_pack_frag(const float* __restrict__ W, int N, int K,
                            _Float16* __restrict__ dst, int nTiles, int ktOff) {
  const int idx = blockIdx.x * blockDim.x + threadIdx.x;
  const int kTiles = K >> 5;
  const int total = kTiles * nTiles * 512;
  if (idx >= total) return;
  const int e  = idx & 15;
  const int l  = (idx >> 4) & 31;
  const int t  = idx >> 9;
  const int kt = t / nTiles;
  const int nt = t - kt * nTiles;
  const int n  = nt * 16 + (l & 15);
  const int k  = kt * 32 + ((e >> 3) << 4) + ((l >> 4) << 3) + (e & 7);
  const _Float16 val = (n < N) ? (_Float16)W[(size_t)n * K + k] : (_Float16)0.0f;
  dst[((size_t)(ktOff + kt) * nTiles + nt) * 512 + (size_t)l * 16 + e] = val;
}

__global__ void k_bias2(const float* a, const float* b, float* o, int n) {
  const int i = blockIdx.x * blockDim.x + threadIdx.x;
  if (i < n) o[i] = a[i] + b[i];
}

__global__ void k_padbias(const float* a, float* o, int n, int npad) {
  const int i = blockIdx.x * blockDim.x + threadIdx.x;
  if (i < npad) o[i] = (i < n) ? a[i] : 0.0f;
}

// ---------------------------------------------------------------------------
// Host launcher
// ---------------------------------------------------------------------------
extern "C" void kernel_launch(void* const* d_in, const int* in_sizes, int n_in,
                              void* d_out, int out_size, void* d_ws, size_t ws_size,
                              hipStream_t stream) {
  (void)in_sizes; (void)n_in; (void)out_size; (void)ws_size;
  const float* features = (const float*)d_in[0];
  const int*   latexes  = (const int*)d_in[1];
  const float* emb      = (const float*)d_in[2];
  const float* Wenc     = (const float*)d_in[3];
  const float* benc     = (const float*)d_in[4];
  const float* Wdec     = (const float*)d_in[5];
  const float* bdec     = (const float*)d_in[6];
  const float* Wattn    = (const float*)d_in[7];
  const float* battn    = (const float*)d_in[8];
  const float* W1ih     = (const float*)d_in[9];
  const float* W1hh     = (const float*)d_in[10];
  const float* b1ih     = (const float*)d_in[11];
  const float* b1hh     = (const float*)d_in[12];
  const float* W2ih     = (const float*)d_in[13];
  const float* W2hh     = (const float*)d_in[14];
  const float* b2ih     = (const float*)d_in[15];
  const float* b2hh     = (const float*)d_in[16];
  const float* Wh0      = (const float*)d_in[17];
  const float* bh0      = (const float*)d_in[18];
  const float* Wc0      = (const float*)d_in[19];
  const float* bc0      = (const float*)d_in[20];
  // d_in[21]=Wg, d_in[22]=bg are dead code in the reference.
  const float* Wf       = (const float*)d_in[23];
  const float* bf       = (const float*)d_in[24];

  char* wsp = (char*)d_ws;
  auto alloc = [&](size_t bytes) -> char* {
    char* p = wsp;
    wsp += (bytes + 255) & ~(size_t)255;
    return p;
  };
  float* enc_t    = (float*)alloc((size_t)128 * 196 * 512 * 4);  // 51.4 MB (L2-resident)
  float* mean_enc = (float*)alloc((size_t)128 * 1024 * 4);
  float* dec_t    = (float*)alloc((size_t)128 * 512 * 4);
  float* xea      = (float*)alloc((size_t)128 * 1152 * 4);       // [emb | awe]
  float* H1a      = (float*)alloc((size_t)128 * 512 * 4);
  float* H1b      = (float*)alloc((size_t)128 * 512 * 4);
  float* H2a      = (float*)alloc((size_t)128 * 512 * 4);
  float* H2b      = (float*)alloc((size_t)128 * 512 * 4);
  float* c1       = (float*)alloc((size_t)128 * 512 * 4);
  float* c2       = (float*)alloc((size_t)128 * 512 * 4);
  _Float16* Wcat1 = (_Float16*)alloc((size_t)52 * 128 * 512 * 2);  // [x1|h1] 1664x2048
  _Float16* Wcat2 = (_Float16*)alloc((size_t)32 * 128 * 512 * 2);  // [h1n|h2] 1024x2048
  _Float16* WencP = (_Float16*)alloc((size_t)32 * 32 * 512 * 2);   // 1024x512
  _Float16* WdecP = (_Float16*)alloc((size_t)16 * 32 * 512 * 2);   // 512x512
  _Float16* WfP   = (_Float16*)alloc((size_t)16 * 63 * 512 * 2);   // 512x1008 (padded)
  float* bsum1    = (float*)alloc(2048 * 4);
  float* bsum2    = (float*)alloc(2048 * 4);
  float* bfpad    = (float*)alloc(1008 * 4);

  float* H1[2] = {H1a, H1b};
  float* H2[2] = {H2a, H2b};
  float* preds  = (float*)d_out;                                  // [128,256,1000]
  float* alphas = (float*)d_out + (size_t)128 * 256 * 1000;       // [128,256,196]

  auto pack = [&](const float* W, int N, int K, _Float16* dst, int nTiles, int ktOff) {
    const int total = (K >> 5) * nTiles * 512;
    k_pack_frag<<<(total + 255) / 256, 256, 0, stream>>>(W, N, K, dst, nTiles, ktOff);
  };

  // -------- setup (runs every call; deterministic) --------
  pack(Wenc, 512, 1024, WencP, 32, 0);
  pack(Wdec, 512, 512, WdecP, 32, 0);
  pack(W1ih, 2048, 1152, Wcat1, 128, 0);
  pack(W1hh, 2048, 512, Wcat1, 128, 36);   // rows 1152.. = 36 k-tiles offset
  pack(W2ih, 2048, 512, Wcat2, 128, 0);
  pack(W2hh, 2048, 512, Wcat2, 128, 16);   // rows 512.. = 16 k-tiles offset
  pack(Wf, 1000, 512, WfP, 63, 0);         // cols 1000..1007 zero-filled
  k_bias2<<<8, 256, 0, stream>>>(b1ih, b1hh, bsum1, 2048);
  k_bias2<<<8, 256, 0, stream>>>(b2ih, b2hh, bsum2, 2048);
  k_padbias<<<4, 256, 0, stream>>>(bf, bfpad, 1000, 1008);
  k_mean<<<128, 256, 0, stream>>>(features, mean_enc);
  k_init_state<<<128, 256, 0, stream>>>(mean_enc, Wh0, bh0, Wc0, bc0,
                                        H1[0], c1, H2[0], c2);
  // enc_t = features @ Wenc.T + benc  (one-time, hoisted; dual-M tiles)
  k_gemm_wmma2<<<dim3(32, 784), 32, 0, stream>>>(features, 1024, WencP, 32,
                                                 benc, enc_t, 512, 1024);

  // -------- 256 sequential decode steps --------
  for (int s = 0; s < 256; ++s) {
    const float* h1cur = H1[s & 1];
    float*       h1nxt = H1[(s + 1) & 1];
    const float* h2cur = H2[s & 1];
    float*       h2nxt = H2[(s + 1) & 1];

    // dec_t = h2 @ Wdec.T + bdec (1 tile/wave: keep 256 waves in flight)
    k_gemm_wmma<<<dim3(32, 8), 32, 0, stream>>>(h2cur, 512, WdecP, 32,
                                                bdec, dec_t, 512, 512);
    // attention + fused embedding gather
    k_attention<<<128, 256, 0, stream>>>(enc_t, dec_t, features, Wattn, battn,
                                         emb, latexes, s,
                                         alphas + (size_t)s * 196, (size_t)256 * 196,
                                         xea);
    // LSTM1: gates = [emb|awe] @ W1ih.T + h1 @ W1hh.T + b
    k_lstm_step<<<dim3(32, 8), 32, 0, stream>>>(xea, 1152, 1152, h1cur, 512, 512,
                                                Wcat1, bsum1, c1, h1nxt);
    // LSTM2: gates = h1n @ W2ih.T + h2 @ W2hh.T + b
    k_lstm_step<<<dim3(32, 8), 32, 0, stream>>>(h1nxt, 512, 512, h2cur, 512, 512,
                                                Wcat2, bsum2, c2, h2nxt);
    // logits = h2n @ Wf.T + bf  -> preds[:, s, :]  (dual-M tiles)
    k_gemm_logits<<<dim3(63, 4), 32, 0, stream>>>(h2nxt, 512, WfP, 63, bfpad,
                                                  preds + (size_t)s * 1000,
                                                  (size_t)256 * 1000, 512, 1000);
  }
}
